// DownVSSBlock_34059090657710
// MI455X (gfx1250) — compile-verified, matrix operand order checked by reading, not run on hardware
//
#include <hip/hip_runtime.h>
#include <hip/hip_bf16.h>
#include <math.h>

typedef __attribute__((ext_vector_type(16))) __bf16 v16bf;
typedef __attribute__((ext_vector_type(8)))  float  v8f;

#define D_MODEL 192
#define D_INNER 384
#define NSTATE  16
#define DTRANK  12
#define KDIR    4
#define BATCHN  4
#define HH      32
#define WW      32
#define LL      1024
#define MROWS   (BATCHN*LL)   /* 4096 pixel rows */
#define YLD     48            /* padded 44 -> 48 */

// ---------------------------------------------------------------------------
// WMMA bf16 operand loaders. CDNA5 16-bit A-matrix 16x32 layout:
//   lanes 0-15:  row = lane,     K in {kb..kb+7, kb+16..kb+23}, kb = k0
//   lanes 16-31: row = lane-16,  same with kb = k0+8
// element e (0..15): k = kb + (e&8 ? 16 : 0) + (e&7)
// => two contiguous 8-float runs per lane: [kb, kb+8) and [kb+16, kb+24)
// ---------------------------------------------------------------------------

// Full-K path: K multiple of 32, no bounds checks -> 4x global_load_b128
__device__ __forceinline__ v16bf load_opnd_full(const float* __restrict__ p, int kb) {
  const float4 a0 = *(const float4*)(p + kb);
  const float4 a1 = *(const float4*)(p + kb + 4);
  const float4 b0 = *(const float4*)(p + kb + 16);
  const float4 b1 = *(const float4*)(p + kb + 20);
  v16bf o;
  o[0]  = (__bf16)a0.x; o[1]  = (__bf16)a0.y; o[2]  = (__bf16)a0.z; o[3]  = (__bf16)a0.w;
  o[4]  = (__bf16)a1.x; o[5]  = (__bf16)a1.y; o[6]  = (__bf16)a1.z; o[7]  = (__bf16)a1.w;
  o[8]  = (__bf16)b0.x; o[9]  = (__bf16)b0.y; o[10] = (__bf16)b0.z; o[11] = (__bf16)b0.w;
  o[12] = (__bf16)b1.x; o[13] = (__bf16)b1.y; o[14] = (__bf16)b1.z; o[15] = (__bf16)b1.w;
  return o;
}

// Padded-K path: clamp the ADDRESS (always valid) and zero the VALUE with a
// select -> plain loads + v_cndmask, no exec-mask branching.
__device__ __forceinline__ v16bf load_opnd_pad(const float* __restrict__ p, int kb, int kmax) {
  v16bf o;
#pragma unroll
  for (int e = 0; e < 16; ++e) {
    int k  = kb + ((e & 8) ? 16 : 0) + (e & 7);
    int kc = (k < kmax) ? k : 0;            // in-bounds address
    float v = p[kc];
    o[e] = (__bf16)((k < kmax) ? v : 0.0f); // cndmask zero-fill
  }
  return o;
}

// ---------------------------------------------------------------------------
// Generic tiled GEMM: C[M,N] = A[M,K] * W[N,K]^T (+ compile-time residual)
// One wave computes a 16x16 f32 tile via v_wmma_f32_16x16x32_bf16.
// blockIdx.y batches over direction k (strided A/W/C).
// ---------------------------------------------------------------------------
template <bool PADK, bool RESID>
__global__ void gemm_bf16_wmma(const float* __restrict__ A, long a_bs, int lda,
                               const float* __restrict__ W, long w_bs, int ldw, int wrows,
                               float* __restrict__ C, long c_bs, int ldc,
                               int M, int N, int Ksz, int Kmax, int nvalid,
                               const float* __restrict__ addsrc, int ldadd) {
  int lane = threadIdx.x & 31;
  int wave = blockIdx.x * (blockDim.x >> 5) + (threadIdx.x >> 5);
  int tilesN = (N + 15) >> 4;
  int tilesM = M >> 4;
  if (wave >= tilesM * tilesN) return;
  int tm = (wave / tilesN) << 4;
  int tn = (wave % tilesN) << 4;
  int half = lane >> 4;
  int r16  = lane & 15;
  int arow = tm + r16;                       // < M always (M multiple of 16)
  int wrow = tn + r16;
  if (wrow >= wrows) wrow = wrows - 1;       // branchless clamp; store guard discards
  const float* Ap = A + (long)blockIdx.y * a_bs + (long)arow * lda;
  const float* Wp = W + (long)blockIdx.y * w_bs + (long)wrow * ldw;
  v8f acc = {0.f, 0.f, 0.f, 0.f, 0.f, 0.f, 0.f, 0.f};
  for (int k0 = 0; k0 < Ksz; k0 += 32) {
    int kb = k0 + half * 8;
    v16bf a, b;
    if (PADK) {
      a = load_opnd_pad(Ap, kb, Kmax);
      b = load_opnd_pad(Wp, kb, Kmax);
    } else {
      a = load_opnd_full(Ap, kb);
      b = load_opnd_full(Wp, kb);
    }
    acc = __builtin_amdgcn_wmma_f32_16x16x32_bf16(
        /*neg_a=*/false, a, /*neg_b=*/false, b,
        /*c_mod=*/(short)0, acc, /*reuse_a=*/false, /*reuse_b=*/false);
  }
  // C/D layout: lane -> N = lane&15, VGPR r -> M = r + 8*(lane>>4)
  int n = tn + r16;
  if (n < nvalid) {
    int m0 = tm + half * 8;
    float* cp = C + (long)blockIdx.y * c_bs + (long)m0 * ldc + n;
    const float* rp = RESID ? (addsrc + (long)m0 * ldadd + n) : nullptr;
#pragma unroll
    for (int r = 0; r < 8; ++r) {
      float v = acc[r];
      if (RESID) v += rp[r * ldadd];
      cp[r * ldc] = v;
    }
  }
}

// ---------------------------------------------------------------------------
// 3x3 stride-2 conv (NHWC, 96 -> 192), pad 1
// ---------------------------------------------------------------------------
__global__ void conv_down(const float* __restrict__ in, const float* __restrict__ cw,
                          const float* __restrict__ cb, float* __restrict__ x) {
  int idx = blockIdx.x * blockDim.x + threadIdx.x;
  if (idx >= MROWS * D_MODEL) return;
  int c   = idx % D_MODEL;
  int row = idx / D_MODEL;
  int b  = row >> 10;
  int oh = (row >> 5) & 31;
  int ow = row & 31;
  float acc = cb[c];
  for (int dy = 0; dy < 3; ++dy) {
    int ih = oh * 2 + dy - 1;
    if (ih < 0 || ih >= 64) continue;
    for (int dx = 0; dx < 3; ++dx) {
      int iw = ow * 2 + dx - 1;
      if (iw < 0 || iw >= 64) continue;
      const float* ip = in + (((long)(b * 64 + ih)) * 64 + iw) * 96;
      const float* wp = cw + (long)((dy * 3 + dx) * 96) * D_MODEL + c;
      for (int ci = 0; ci < 96; ++ci)
        acc += ip[ci] * wp[(long)ci * D_MODEL];
    }
  }
  x[idx] = acc;
}

// ---------------------------------------------------------------------------
// LayerNorm over 192 channels (one wave per pixel), eps 1e-6
// ---------------------------------------------------------------------------
__global__ void layernorm192(const float* __restrict__ x, const float* __restrict__ g,
                             const float* __restrict__ bta, float* __restrict__ xn) {
  int lane = threadIdx.x & 31;
  int pix  = blockIdx.x * (blockDim.x >> 5) + (threadIdx.x >> 5);
  if (pix >= MROWS) return;
  const float* p = x + (long)pix * D_MODEL;
  float v[6], s = 0.f, s2 = 0.f;
#pragma unroll
  for (int i = 0; i < 6; ++i) { v[i] = p[lane + 32 * i]; s += v[i]; s2 += v[i] * v[i]; }
  for (int o = 16; o > 0; o >>= 1) { s += __shfl_xor(s, o, 32); s2 += __shfl_xor(s2, o, 32); }
  float mu  = s / 192.f;
  float var = s2 / 192.f - mu * mu;
  float rs  = rsqrtf(var + 1e-6f);
  float* q = xn + (long)pix * D_MODEL;
#pragma unroll
  for (int i = 0; i < 6; ++i) { int c = lane + 32 * i; q[c] = (v[i] - mu) * rs * g[c] + bta[c]; }
}

// ---------------------------------------------------------------------------
// Depthwise 3x3 (pad 1) on xc = xz[:, :384] + bias + SiLU -> xca (4096,384)
// ---------------------------------------------------------------------------
__global__ void dwconv_silu(const float* __restrict__ xz, const float* __restrict__ w,
                            const float* __restrict__ bias, float* __restrict__ xca) {
  int idx = blockIdx.x * blockDim.x + threadIdx.x;
  if (idx >= MROWS * D_INNER) return;
  int d   = idx % D_INNER;
  int row = idx / D_INNER;
  int b = row >> 10, h = (row >> 5) & 31, wc = row & 31;
  float acc = bias[d];
#pragma unroll
  for (int dy = 0; dy < 3; ++dy) {
    int ih = h + dy - 1;
    if (ih < 0 || ih >= HH) continue;
#pragma unroll
    for (int dx = 0; dx < 3; ++dx) {
      int iw = wc + dx - 1;
      if (iw < 0 || iw >= WW) continue;
      acc += xz[((long)(b * 1024 + ih * 32 + iw)) * 768 + d] * w[(dy * 3 + dx) * D_INNER + d];
    }
  }
  acc = acc / (1.f + __expf(-acc));   // SiLU
  xca[idx] = acc;
}

// ---------------------------------------------------------------------------
// Selective scan. One wave per (b, k, 32-channel group); lane = channel.
// Direction permutation applied branchlessly in address math; B/C (32 shared
// floats per step) loaded coalesced one-per-lane, broadcast via ds_bpermute.
// h[n] = exp(dt*A[n])*h[n] + dt*u*B[n];  y = sum_n h[n]*C[n]
// ---------------------------------------------------------------------------
__global__ void selective_scan_k(const float* __restrict__ xca, const float* __restrict__ Y,
                                 const float* __restrict__ dtp, const float* __restrict__ A_logs,
                                 const float* __restrict__ dt_b, float* __restrict__ ydir) {
  int lane = threadIdx.x & 31;
  int wave = blockIdx.x * (blockDim.x >> 5) + (threadIdx.x >> 5);
  if (wave >= BATCHN * KDIR * (D_INNER / 32)) return;
  int dg = wave % (D_INNER / 32);
  int k  = (wave / (D_INNER / 32)) % KDIR;
  int b  = wave / ((D_INNER / 32) * KDIR);
  int d  = dg * 32 + lane;
  float Ac[NSTATE];
#pragma unroll
  for (int n = 0; n < NSTATE; ++n)
    Ac[n] = -__expf(A_logs[((long)k * D_INNER + d) * NSTATE + n]);   // A = -exp(A_log)
  float dtb = dt_b[k * D_INNER + d];
  float h[NSTATE];
#pragma unroll
  for (int n = 0; n < NSTATE; ++n) h[n] = 0.f;
  const int flip = (k & 2) ? 1 : 0;     // wave-uniform
  const int swap = (k & 1) ? 1 : 0;
  for (int t = 0; t < LL; ++t) {
    int t2 = flip ? (LL - 1 - t) : t;
    int s  = swap ? (((t2 & 31) << 5) | (t2 >> 5)) : t2;   // (h,w)<->(w,h)
    long row  = (long)b * LL + s;
    long krow = (long)k * MROWS + row;
    float u    = xca[row * D_INNER + d];
    float dpre = dtp[krow * D_INNER + d] + dtb;
    float dt   = (dpre > 20.f) ? dpre : log1pf(__expf(dpre));        // softplus
    float ybc  = Y[krow * YLD + DTRANK + lane];                      // lanes 0-15: B, 16-31: C
    float du   = dt * u;
    float y    = 0.f;
#pragma unroll
    for (int n = 0; n < NSTATE; ++n) {
      float bn = __shfl(ybc, n, 32);
      float cn = __shfl(ybc, n + 16, 32);
      h[n] = __expf(dt * Ac[n]) * h[n] + du * bn;
      y   += h[n] * cn;
    }
    ydir[krow * D_INNER + d] = y;    // output lands at same spatial index it read
  }
}

// ---------------------------------------------------------------------------
// y = sum_k ydir_k + (sum_k Ds[k,:]) * xca ; LayerNorm(384, eps 1e-5);
// gate with SiLU(z) ; write gbuf for out_proj GEMM. One wave per pixel.
// ---------------------------------------------------------------------------
__global__ void combine_ln_gate(const float* __restrict__ ydir, const float* __restrict__ xca,
                                const float* __restrict__ Ds, const float* __restrict__ xz,
                                const float* __restrict__ g, const float* __restrict__ bta,
                                float* __restrict__ gout) {
  int lane = threadIdx.x & 31;
  int pix  = blockIdx.x * (blockDim.x >> 5) + (threadIdx.x >> 5);
  if (pix >= MROWS) return;
  float v[12], s = 0.f, s2 = 0.f;
#pragma unroll
  for (int i = 0; i < 12; ++i) {
    int d = lane + 32 * i;
    float u    = xca[(long)pix * D_INNER + d];
    float dsum = Ds[d] + Ds[D_INNER + d] + Ds[2 * D_INNER + d] + Ds[3 * D_INNER + d];
    float y = dsum * u;
#pragma unroll
    for (int k = 0; k < KDIR; ++k) y += ydir[((long)k * MROWS + pix) * D_INNER + d];
    v[i] = y; s += y; s2 += y * y;
  }
  for (int o = 16; o > 0; o >>= 1) { s += __shfl_xor(s, o, 32); s2 += __shfl_xor(s2, o, 32); }
  float mu = s / 384.f, var = s2 / 384.f - mu * mu, rs = rsqrtf(var + 1e-5f);
#pragma unroll
  for (int i = 0; i < 12; ++i) {
    int d = lane + 32 * i;
    float z  = xz[(long)pix * 768 + D_INNER + d];
    float sz = z / (1.f + __expf(-z));
    gout[(long)pix * D_INNER + d] = ((v[i] - mu) * rs * g[d] + bta[d]) * sz;
  }
}

// ---------------------------------------------------------------------------
extern "C" void kernel_launch(void* const* d_in, const int* in_sizes, int n_in,
                              void* d_out, int out_size, void* d_ws, size_t ws_size,
                              hipStream_t stream) {
  const float* inputs     = (const float*)d_in[0];
  const float* conv_w     = (const float*)d_in[1];
  const float* conv_b     = (const float*)d_in[2];
  const float* in_proj_w  = (const float*)d_in[3];
  const float* dw_w       = (const float*)d_in[4];
  const float* dw_b       = (const float*)d_in[5];
  const float* x_proj_w   = (const float*)d_in[6];
  const float* dt_w       = (const float*)d_in[7];
  const float* dt_b       = (const float*)d_in[8];
  const float* A_logs     = (const float*)d_in[9];
  const float* Ds         = (const float*)d_in[10];
  const float* onorm_g    = (const float*)d_in[11];
  const float* onorm_b    = (const float*)d_in[12];
  const float* out_proj_w = (const float*)d_in[13];
  const float* ln1_g      = (const float*)d_in[14];
  const float* ln1_b      = (const float*)d_in[15];
  float* out = (float*)d_out;

  float* ws   = (float*)d_ws;
  float* x    = ws;                          // 4096*192  residual
  float* xn   = x    + (long)MROWS * D_MODEL;
  float* xz   = xn   + (long)MROWS * D_MODEL;            // 4096*768
  float* xca  = xz   + (long)MROWS * 2 * D_INNER;        // 4096*384
  float* Y    = xca  + (long)MROWS * D_INNER;            // 4*4096*48  (dt|B|C per dir)
  float* dtp  = Y    + (long)KDIR * MROWS * YLD;         // 4*4096*384
  float* ydir = dtp  + (long)KDIR * MROWS * D_INNER;     // 4*4096*384
  float* gbuf = ydir + (long)KDIR * MROWS * D_INNER;     // 4096*384

  // 1. stride-2 downsample conv
  conv_down<<<(MROWS * D_MODEL + 255) / 256, 256, 0, stream>>>(inputs, conv_w, conv_b, x);
  // 2. LN (eps 1e-6)
  layernorm192<<<MROWS / 8, 256, 0, stream>>>(x, ln1_g, ln1_b, xn);
  // 3. in_proj: (4096x192)@(192->768)
  {
    int tiles = (MROWS / 16) * (768 / 16);
    gemm_bf16_wmma<false, false><<<dim3((tiles + 7) / 8, 1), 256, 0, stream>>>(
        xn, 0, D_MODEL, in_proj_w, 0, D_MODEL, 768, xz, 0, 768,
        MROWS, 768, D_MODEL, D_MODEL, 768, nullptr, 0);
  }
  // 4. depthwise conv + SiLU
  dwconv_silu<<<(MROWS * D_INNER + 255) / 256, 256, 0, stream>>>(xz, dw_w, dw_b, xca);
  // 5. x_proj per direction (un-permuted; perm folded into scan addressing)
  {
    int tiles = (MROWS / 16) * (YLD / 16);
    gemm_bf16_wmma<false, false><<<dim3((tiles + 7) / 8, KDIR), 256, 0, stream>>>(
        xca, 0, D_INNER, x_proj_w, (long)44 * D_INNER, D_INNER, 44,
        Y, (long)MROWS * YLD, YLD,
        MROWS, YLD, D_INNER, D_INNER, 44, nullptr, 0);
  }
  // 6. dt projection per direction: K=12 zero-padded to one 32-wide WMMA step
  {
    int tiles = (MROWS / 16) * (D_INNER / 16);
    gemm_bf16_wmma<true, false><<<dim3((tiles + 7) / 8, KDIR), 256, 0, stream>>>(
        Y, (long)MROWS * YLD, YLD, dt_w, (long)D_INNER * DTRANK, DTRANK, D_INNER,
        dtp, (long)MROWS * D_INNER, D_INNER,
        MROWS, D_INNER, 32, DTRANK, D_INNER, nullptr, 0);
  }
  // 7. the serial scan (192 waves, h[16] in registers)
  selective_scan_k<<<(BATCHN * KDIR * (D_INNER / 32) * 32 + 255) / 256, 256, 0, stream>>>(
      xca, Y, dtp, A_logs, dt_b, ydir);
  // 8. combine 4 directions + D*u, LN(384, eps 1e-5), SiLU(z) gate
  combine_ln_gate<<<MROWS / 8, 256, 0, stream>>>(ydir, xca, Ds, xz, onorm_g, onorm_b, gbuf);
  // 9. out_proj + residual
  {
    int tiles = (MROWS / 16) * (D_MODEL / 16);
    gemm_bf16_wmma<false, true><<<dim3((tiles + 7) / 8, 1), 256, 0, stream>>>(
        gbuf, 0, D_INNER, out_proj_w, 0, D_INNER, D_MODEL, out, 0, D_MODEL,
        MROWS, D_MODEL, D_INNER, D_INNER, D_MODEL, x, D_MODEL);
  }
}
